// VxVDFN_25812753449762
// MI455X (gfx1250) — compile-verified, matrix operand order checked by reading, not run on hardware
//
#include <hip/hip_runtime.h>
#include <hip/hip_bf16.h>

typedef __attribute__((ext_vector_type(16))) __bf16 v16bf;
typedef __attribute__((ext_vector_type(8)))  float  v8f;

#define Cc 64
#define Hh 160
#define Ww 160
#define TP 32          // pixels (along W) per block
#define HALO 2         // union halo of dilation-1 and dilation-2 3x3
#define TR (2*HALO+1)  // 5 staged rows
#define TC (TP+2*HALO) // 36 staged cols
// B fragment workspace: [br][nt][ks][lane][16] bf16
#define NFRAG_ELEMS (2*36*2*32*16)   // 73728 bf16 = 147456 bytes

union BF8 { uint4 u; __bf16 b[8]; };

// ---------------- prologue: pack pw_w (f32) into bf16 WMMA B fragments ----------------
__global__ __launch_bounds__(256)
void pack_b_frags(const float* __restrict__ pw0, const float* __restrict__ pw1,
                  __bf16* __restrict__ frag)
{
    int id = blockIdx.x * 256 + threadIdx.x;
    if (id >= NFRAG_ELEMS) return;
    int i    = id & 15;
    int lane = (id >> 4) & 31;
    int ks   = (id >> 9) & 1;
    int q    = id >> 10;            // (br*36 + nt)
    int nt   = q % 36;
    int br   = q / 36;
    const float* pw = br ? pw1 : pw0;
    int half = lane >> 4;
    int n    = lane & 15;
    int o    = nt * 16 + n;
    int k    = ks * 32 + half * 8 + ((i < 8) ? i : 16 + (i - 8));
    frag[id] = (__bf16)pw[o * Cc + k];
}

// ---------------- main fused kernel ----------------
template <bool USE_FRAG>
__global__ __launch_bounds__(256)
void ddf_fused_wmma(const float* __restrict__ y, const float* __restrict__ x,
                    const float* __restrict__ dw_w0, const float* __restrict__ bn_g0,
                    const float* __restrict__ bn_b0, const float* __restrict__ bn_m0,
                    const float* __restrict__ bn_v0, const float* __restrict__ pw_w0,
                    const float* __restrict__ pw_b0,
                    const float* __restrict__ dw_w1, const float* __restrict__ bn_g1,
                    const float* __restrict__ bn_b1, const float* __restrict__ bn_m1,
                    const float* __restrict__ bn_v1, const float* __restrict__ pw_w1,
                    const float* __restrict__ pw_b1,
                    const float* __restrict__ norm_g, const float* __restrict__ norm_b,
                    const float* __restrict__ norm_m, const float* __restrict__ norm_v,
                    const __bf16* __restrict__ bfrags,
                    float* __restrict__ out)
{
    __shared__ float  tile[Cc * TR * TC];  // staged halo tile: first y, then x (46080 B)
    __shared__ __bf16 tA[2][TP * Cc];      // bf16 A operands per branch, [m][c]  (8192 B)
    __shared__ float  outAcc[TP * Cc];     // f32 output accumulator, [m][c]      (8192 B)
    __shared__ float  bnSc[2][Cc], bnSh[2][Cc];   // branch BN scale/shift       (1024 B)
    __shared__ float  nrmSc[Cc], nrmSh[Cc];       // final BN scale/shift        (512 B)

    const int tid = threadIdx.x;
    const int blk = blockIdx.x;
    const int wt  = blk % (Ww / TP);
    const int h   = (blk / (Ww / TP)) % Hh;
    const int n   = blk / ((Ww / TP) * Hh);
    const int wbase = wt * TP;

    // per-thread staging sub-patch: channel c0, quadrant covers cols [q*9, q*9+9)
    const int c0 = tid >> 2;
    const int q4 = tid & 3;

    // ---------- stage 0: BN tables, y halo tile -> LDS, zero accumulator ----------
    if (tid < 128) {             // branch BN: 2 x 64 channels
        int c = tid & 63, b = tid >> 6;
        float g  = (b ? bn_g1 : bn_g0)[c];
        float bb = (b ? bn_b1 : bn_b0)[c];
        float mm = (b ? bn_m1 : bn_m0)[c];
        float vv = (b ? bn_v1 : bn_v0)[c];
        float sc = g * rsqrtf(vv + 1e-5f);
        bnSc[b][c] = sc;
        bnSh[b][c] = bb - mm * sc;
    } else if (tid < 192) {      // final BN: 64 channels
        int c = tid - 128;
        float sc = norm_g[c] * rsqrtf(norm_v[c] + 1e-5f);
        nrmSc[c] = sc;
        nrmSh[c] = norm_b[c] - norm_m[c] * sc;
    }
    {
        const float* yc = y + ((size_t)(n * Cc + c0) * Hh) * Ww;
        float* tc = tile + c0 * (TR * TC) + q4 * 9;
        #pragma unroll
        for (int row = 0; row < TR; ++row) {
            int hy = h + row - HALO;
            bool hok = (unsigned)hy < Hh;
            const float* yr = yc + (size_t)hy * Ww;
            #pragma unroll
            for (int j = 0; j < 9; ++j) {
                int wy = wbase + q4 * 9 + j - HALO;
                float v = (hok && (unsigned)wy < Ww) ? yr[wy] : 0.f;
                tc[row * TC + j] = v;
            }
        }
        #pragma unroll
        for (int ii = 0; ii < TP * Cc / 256; ++ii) outAcc[tid + 256 * ii] = 0.f;
    }
    __syncthreads();

    // ---------- stage 1: depthwise conv + BN + ReLU from LDS -> bf16 A tiles ----------
    for (int ii = 0; ii < TP * Cc / 256; ++ii) {
        int idx = tid + 256 * ii;        // (m,c) pairs
        int c = idx & (Cc - 1);
        int m = idx >> 6;
        const float* tc = tile + c * (TR * TC);
        #pragma unroll
        for (int br = 0; br < 2; ++br) {
            const int r = br + 1;        // dilation: 1, 2
            const float* dw = (br ? dw_w1 : dw_w0) + c * 9;
            float acc = 0.f;
            #pragma unroll
            for (int i = 0; i < 3; ++i) {
                #pragma unroll
                for (int j = 0; j < 3; ++j) {
                    acc += tc[(HALO + r * (i - 1)) * TC + (HALO + m + r * (j - 1))]
                           * dw[i * 3 + j];
                }
            }
            float t = acc * bnSc[br][c] + bnSh[br][c];
            t = t > 0.f ? t : 0.f;
            tA[br][m * Cc + c] = (__bf16)t;
        }
    }
    __syncthreads();

    // ---------- stage 2: x halo tile -> LDS (overwrites y tile) ----------
    {
        const float* xc = x + ((size_t)(n * Cc + c0) * Hh) * Ww;
        float* tc = tile + c0 * (TR * TC) + q4 * 9;
        #pragma unroll
        for (int row = 0; row < TR; ++row) {
            int hy = h + row - HALO;
            bool hok = (unsigned)hy < Hh;
            const float* xr = xc + (size_t)hy * Ww;
            #pragma unroll
            for (int j = 0; j < 9; ++j) {
                int wy = wbase + q4 * 9 + j - HALO;
                float v = (hok && (unsigned)wy < Ww) ? xr[wy] : 0.f;
                tc[row * TC + j] = v;
            }
        }
    }
    __syncthreads();

    // ---------- stage 3: per-pixel kernel prediction GEMM via WMMA ----------
    const int lane = tid & 31;
    const int wave = tid >> 5;
    const int br   = wave >> 2;          // waves 0-3: branch0, 4-7: branch1
    const int r    = br + 1;
    const int wgrp = wave & 3;           // 9 N-tiles per wave
    const int mrow = lane & 15;
    const int half = lane >> 4;

    // A fragments: 2 M-tiles x 2 k-steps. 16-bit A layout: lanes 0-15 hold
    // K {0..7,16..23}, lanes 16-31 hold K {8..15,24..31}.
    v16bf afrag[2][2];
    {
        const uint4* tb4 = reinterpret_cast<const uint4*>(tA[br]);
        #pragma unroll
        for (int mt = 0; mt < 2; ++mt) {
            #pragma unroll
            for (int ks = 0; ks < 2; ++ks) {
                int e0 = ((mt * 16 + mrow) * Cc + ks * 32 + half * 8) >> 3;
                BF8 lo, hi;
                lo.u = tb4[e0];
                hi.u = tb4[e0 + 2];                          // +16 elements
                #pragma unroll
                for (int i = 0; i < 8; ++i) {
                    afrag[mt][ks][i]     = lo.b[i];
                    afrag[mt][ks][8 + i] = hi.b[i];
                }
            }
        }
    }

    const float*  pw    = br ? pw_w1 : pw_w0;
    const float*  pb    = br ? pw_b1 : pw_b0;
    const v16bf*  fbase = USE_FRAG
        ? reinterpret_cast<const v16bf*>(bfrags) + ((size_t)br * 36 * 2 * 32 + lane)
        : nullptr;

    #pragma unroll
    for (int t9 = 0; t9 < 9; ++t9) {
        const int nt = wgrp + 4 * t9;          // N-tile 0..35
        const int o  = nt * 16 + mrow;         // output channel (this lane's column)

        v8f dacc[2];
        dacc[0] = v8f{0.f, 0.f, 0.f, 0.f, 0.f, 0.f, 0.f, 0.f};
        dacc[1] = dacc[0];
        #pragma unroll
        for (int ks = 0; ks < 2; ++ks) {
            v16bf bfrag;
            if (USE_FRAG) {
                // one contiguous 32-byte load per lane: pre-packed bf16 fragment
                bfrag = fbase[(size_t)(nt * 2 + ks) * 32];
            } else {
                const float4* row4 =
                    reinterpret_cast<const float4*>(pw + (size_t)o * Cc + ks * 32 + half * 8);
                float4 f0 = row4[0], f1 = row4[1];   // K chunk base+0..7
                float4 f2 = row4[4], f3 = row4[5];   // K chunk base+16..23
                bfrag[0]  = (__bf16)f0.x; bfrag[1]  = (__bf16)f0.y;
                bfrag[2]  = (__bf16)f0.z; bfrag[3]  = (__bf16)f0.w;
                bfrag[4]  = (__bf16)f1.x; bfrag[5]  = (__bf16)f1.y;
                bfrag[6]  = (__bf16)f1.z; bfrag[7]  = (__bf16)f1.w;
                bfrag[8]  = (__bf16)f2.x; bfrag[9]  = (__bf16)f2.y;
                bfrag[10] = (__bf16)f2.z; bfrag[11] = (__bf16)f2.w;
                bfrag[12] = (__bf16)f3.x; bfrag[13] = (__bf16)f3.y;
                bfrag[14] = (__bf16)f3.z; bfrag[15] = (__bf16)f3.w;
            }
            // one B fragment feeds both M-tiles (independent chains interleave)
            dacc[0] = __builtin_amdgcn_wmma_f32_16x16x32_bf16(
                          false, afrag[0][ks], false, bfrag, (short)0, dacc[0], false, false);
            dacc[1] = __builtin_amdgcn_wmma_f32_16x16x32_bf16(
                          false, afrag[1][ks], false, bfrag, (short)0, dacc[1], false, false);
        }

        // epilogue: bias, map o -> (channel c, tap); multiply with staged x (LDS,
        // pre-padded -> no bounds checks, no divergence); accumulate via ds_add_f32.
        const float bias = pb[o];
        const int c   = o / 9;
        const int tap = o - c * 9;
        const int ti  = tap / 3, tj = tap - ti * 3;
        const float* xr = tile + (c * TR + (HALO + r * (ti - 1))) * TC + (HALO + r * (tj - 1));
        #pragma unroll
        for (int mt = 0; mt < 2; ++mt) {
            #pragma unroll
            for (int e = 0; e < 8; ++e) {
                int m = mt * 16 + e + 8 * half;     // pixel row of D element e
                float xv = xr[m];
                float kv = dacc[mt][e] + bias;
                __hip_atomic_fetch_add(&outAcc[m * Cc + c], kv * xv,
                                       __ATOMIC_RELAXED, __HIP_MEMORY_SCOPE_WORKGROUP);
            }
        }
    }
    __syncthreads();

    // ---------- stage 4: final BN + ReLU, coalesced store ----------
    #pragma unroll
    for (int ii = 0; ii < TP * Cc / 256; ++ii) {
        int idx = tid + 256 * ii;
        int m = idx & (TP - 1);                  // consecutive tids -> consecutive w
        int c = idx / TP;
        float v = outAcc[m * Cc + c] * nrmSc[c] + nrmSh[c];
        out[((size_t)(n * Cc + c) * Hh + h) * Ww + wbase + m] = v > 0.f ? v : 0.f;
    }
}

extern "C" void kernel_launch(void* const* d_in, const int* in_sizes, int n_in,
                              void* d_out, int out_size, void* d_ws, size_t ws_size,
                              hipStream_t stream) {
    (void)in_sizes; (void)n_in; (void)out_size;
    const float* y      = (const float*)d_in[0];
    const float* x      = (const float*)d_in[1];
    const float* dw_w0  = (const float*)d_in[2];
    const float* bn_g0  = (const float*)d_in[3];
    const float* bn_b0  = (const float*)d_in[4];
    const float* bn_m0  = (const float*)d_in[5];
    const float* bn_v0  = (const float*)d_in[6];
    const float* pw_w0  = (const float*)d_in[7];
    const float* pw_b0  = (const float*)d_in[8];
    const float* dw_w1  = (const float*)d_in[9];
    const float* bn_g1  = (const float*)d_in[10];
    const float* bn_b1  = (const float*)d_in[11];
    const float* bn_m1  = (const float*)d_in[12];
    const float* bn_v1  = (const float*)d_in[13];
    const float* pw_w1  = (const float*)d_in[14];
    const float* pw_b1  = (const float*)d_in[15];
    const float* norm_g = (const float*)d_in[16];
    const float* norm_b = (const float*)d_in[17];
    const float* norm_m = (const float*)d_in[18];
    const float* norm_v = (const float*)d_in[19];
    float* outp = (float*)d_out;

    const int blocks = 4 * Hh * (Ww / TP);   // N * H * (W/32) = 3200

    if (ws_size >= (size_t)NFRAG_ELEMS * sizeof(__bf16)) {
        __bf16* frag = (__bf16*)d_ws;
        pack_b_frags<<<(NFRAG_ELEMS + 255) / 256, 256, 0, stream>>>(pw_w0, pw_w1, frag);
        ddf_fused_wmma<true><<<blocks, 256, 0, stream>>>(
            y, x,
            dw_w0, bn_g0, bn_b0, bn_m0, bn_v0, pw_w0, pw_b0,
            dw_w1, bn_g1, bn_b1, bn_m1, bn_v1, pw_w1, pw_b1,
            norm_g, norm_b, norm_m, norm_v, frag, outp);
    } else {
        ddf_fused_wmma<false><<<blocks, 256, 0, stream>>>(
            y, x,
            dw_w0, bn_g0, bn_b0, bn_m0, bn_v0, pw_w0, pw_b0,
            dw_w1, bn_g1, bn_b1, bn_m1, bn_v1, pw_w1, pw_b1,
            norm_g, norm_b, norm_m, norm_v, nullptr, outp);
    }
}
